// EdgeAwareNodeBlock_18683107737868
// MI455X (gfx1250) — matrix operand order repaired and checked by reading, not compiled, not run on hardware
//
#include <hip/hip_runtime.h>
#include <hip/hip_bf16.h>
#include <math.h>

#define BB 8
#define NN 256
#define DD 128
#define EPS_AGG 1e-6f
#define LN_EPS  1e-5f

typedef _Float16 v16h __attribute__((ext_vector_type(16)));
typedef _Float16 v8h  __attribute__((ext_vector_type(8)));
typedef float    v8f  __attribute__((ext_vector_type(8)));
typedef float    f32x4 __attribute__((ext_vector_type(4)));

// ---------------------------------------------------------------------------
// Phase 1: per-node precompute
//   Apre[n,d] = sum_k x[n,k]*W1m[k,d]        + b1m[d]   (xi contribution + bias)
//   Bp  [n,d] = sum_k x[n,k]*W1m[128+k,d]               (xj contribution)
// ---------------------------------------------------------------------------
__global__ void ean_phase1(const float* __restrict__ x,
                           const float* __restrict__ W1m,
                           const float* __restrict__ b1m,
                           float* __restrict__ Apre,
                           float* __restrict__ Bp)
{
    const int node = blockIdx.x;          // 0 .. B*N-1
    const int t    = threadIdx.x;         // 0 .. 127  (= output column d)
    __shared__ float xr[DD];
    xr[t] = x[(size_t)node * DD + t];
    __syncthreads();

    float a = b1m[t];
    float p = 0.0f;
#pragma unroll 8
    for (int k = 0; k < DD; ++k) {
        const float xv = xr[k];
        a = fmaf(xv, W1m[(size_t)k * DD + t], a);
        p = fmaf(xv, W1m[(size_t)(DD + k) * DD + t], p);
    }
    Apre[(size_t)node * DD + t] = a;
    Bp  [(size_t)node * DD + t] = p;
}

// ---------------------------------------------------------------------------
// Phase 2: dominant WMMA kernel. One workgroup (256 threads / 8 waves) per
// node (b,i). Computes hsum[b,i,d] = sum_j mask * relu(Apre[i] + Bp[j]
//                                                      + edge0[b,i,j,:]@W1m_c)
// ---------------------------------------------------------------------------
#define E_STRIDE 136   // halves per LDS row (128 + 8 pad -> 272B, 16B aligned, conflict-free)

__global__ void __launch_bounds__(256)
ean_phase2(const float* __restrict__ edge0,
           const int*   __restrict__ edge_mask,
           const float* __restrict__ W1m,      // rows 256..383 = W1m_c
           const float* __restrict__ Apre,
           const float* __restrict__ Bp,
           float* __restrict__ hsum)
{
    const int node = blockIdx.x;              // b*N + i
    const int b    = node >> 8;               // node / N
    const int t    = threadIdx.x;
    const int wave = t >> 5;                  // 0..7  -> d-tile
    const int lane = t & 31;
    const int laneLo = lane & 15;
    const int hi   = (lane >> 4) & 1;         // lane 16..31 -> 1
    const int d0   = wave * 16;
    const int col  = d0 + laneLo;             // output column this lane owns

    __shared__ __align__(16) _Float16 sE[16 * E_STRIDE];  // f16 edge tile (16 x 128)
    __shared__ __align__(16) float    sBp[16 * DD];       // Bp tile (16 x 128)
    __shared__ float                  sMask[NN];          // mask row as float

    // --- stage mask row once ---
    sMask[t] = (float)edge_mask[(size_t)node * NN + t];

    // --- build B fragments of W1m_c[:, d0:d0+16] (held in VGPRs all kernel) ---
    // B 32x16 in this code's contiguous k order (same as A): lanes 0-15 col=lane hold K=0..15 (2 per VGPR),
    //                        lanes 16-31 col=lane-16 hold K=16..31.
    v16h bfrag[4];
#pragma unroll
    for (int kk = 0; kk < 4; ++kk) {
#pragma unroll
        for (int h = 0; h < 16; ++h) {
            const int k = kk * 32 + hi * 16 + h;
            bfrag[kk][h] = (_Float16)W1m[(size_t)(2 * DD + k) * DD + col];
        }
    }

    const float cbias = Apre[(size_t)node * DD + col];   // Apre already includes b1m

    // cooperative-load geometry (16 rows x 128 cols, 8 elems / thread)
    const int row = t >> 4;          // 0..15
    const int c0  = (t & 15) * 8;    // 0,8,...,120

    float acc = 0.0f;                // running masked column sum

    for (int jt = 0; jt < 16; ++jt) {
        const int j0 = jt * 16;

        // ---- stage edge tile (f32 -> f16) and Bp tile into LDS ----
        const float* ep = edge0 + (((size_t)node * NN + j0 + row) * DD + c0);
        f32x4 e0 = *(const f32x4*)ep;
        f32x4 e1 = *(const f32x4*)(ep + 4);
        _Float16* de = &sE[row * E_STRIDE + c0];
#pragma unroll
        for (int q = 0; q < 4; ++q) { de[q] = (_Float16)e0[q]; de[4 + q] = (_Float16)e1[q]; }

        const float* bpp = Bp + (((size_t)b * NN + j0 + row) * DD + c0);
        *(f32x4*)&sBp[row * DD + c0]       = *(const f32x4*)bpp;
        *(f32x4*)&sBp[row * DD + c0 + 4]   = *(const f32x4*)(bpp + 4);

        if (jt < 15) // hint next tile into cache while we compute
            __builtin_prefetch(edge0 + (((size_t)node * NN + j0 + 16 + row) * DD + c0), 0, 0);

        __syncthreads();

        // ---- 16x16 (j x d) tile of H = E @ W1m_c via 4 K-chunks of 32 ----
        v8f cacc = {};
#pragma unroll
        for (int kk = 0; kk < 4; ++kk) {
            // A 16x32 in this code's contiguous k order (same as B): lanes 0-15 row=lane, halves = K{0..15};
            //                        lanes 16-31 row=lane-16, halves = K{16..31}
            const int hbase = laneLo * E_STRIDE + kk * 32 + hi * 16;
            v8h a0 = *(const v8h*)&sE[hbase];
            v8h a1 = *(const v8h*)&sE[hbase + 8];
            v16h afrag = __builtin_shufflevector(a0, a1,
                              0, 1, 2, 3, 4, 5, 6, 7, 8, 9, 10, 11, 12, 13, 14, 15);
            cacc = __builtin_amdgcn_wmma_f32_16x16x32_f16(
                       false, afrag, false, bfrag[kk], (short)0, cacc, false, false);
        }

        // ---- bias + Bp + ReLU + mask, reduce 8 rows into column accumulator ----
#pragma unroll
        for (int r = 0; r < 8; ++r) {
            const int m  = r + hi * 8;                 // row within tile this VGPR holds
            float v = cacc[r] + cbias + sBp[m * DD + col];
            v = v > 0.0f ? v : 0.0f;
            acc = fmaf(v, sMask[j0 + m], acc);
        }
        __syncthreads();   // before next iteration overwrites sE/sBp
    }

    // combine the two row-halves (lane L holds rows 0-7 sum, L+16 rows 8-15 sum)
    acc += __shfl_xor(acc, 16, 32);
    if (lane < 16)
        hsum[(size_t)node * DD + col] = acc;
}

// ---------------------------------------------------------------------------
// Phase 3: per-node tail:  agg = (hsum@W2m + cnt*b2m)/(cnt+eps);
// update MLP, residual, LayerNorm.
// ---------------------------------------------------------------------------
__global__ void ean_phase3(const float* __restrict__ x,
                           const int*   __restrict__ edge_mask,
                           const float* __restrict__ hsum,
                           const float* __restrict__ W2m, const float* __restrict__ b2m,
                           const float* __restrict__ W1u, const float* __restrict__ b1u,
                           const float* __restrict__ W2u, const float* __restrict__ b2u,
                           const float* __restrict__ gamma, const float* __restrict__ beta,
                           float* __restrict__ out)
{
    const int node = blockIdx.x;
    const int t    = threadIdx.x;     // 0..127
    __shared__ float sh[DD];          // hsum row
    __shared__ float red[DD];         // reduction scratch
    __shared__ float uin[2 * DD];     // [x, agg]
    __shared__ float hu[DD];

    // ---- cnt = sum_j mask[b,i,j] ----
    const int* mrow = edge_mask + (size_t)node * NN;
    red[t] = (float)(mrow[t] + mrow[t + DD]);
    __syncthreads();
    for (int s = 64; s > 0; s >>= 1) {
        if (t < s) red[t] += red[t + s];
        __syncthreads();
    }
    const float cnt = red[0];
    __syncthreads();

    const float xval = x[(size_t)node * DD + t];
    sh[t]  = hsum[(size_t)node * DD + t];
    uin[t] = xval;
    __syncthreads();

    // ---- agg = (hsum@W2m + cnt*b2m)/(cnt+eps) ----
    float aggv = cnt * b2m[t];
#pragma unroll 8
    for (int k = 0; k < DD; ++k)
        aggv = fmaf(sh[k], W2m[(size_t)k * DD + t], aggv);
    aggv /= (cnt + EPS_AGG);
    uin[DD + t] = aggv;
    __syncthreads();

    // ---- hu = relu(uin @ W1u + b1u) ----
    float h = b1u[t];
#pragma unroll 8
    for (int k = 0; k < 2 * DD; ++k)
        h = fmaf(uin[k], W1u[(size_t)k * DD + t], h);
    hu[t] = h > 0.0f ? h : 0.0f;
    __syncthreads();

    // ---- out = hu @ W2u + b2u ; y = x + out ----
    float o = b2u[t];
#pragma unroll 8
    for (int k = 0; k < DD; ++k)
        o = fmaf(hu[k], W2u[(size_t)k * DD + t], o);
    const float y = xval + o;

    // ---- LayerNorm ----
    red[t] = y;
    __syncthreads();
    for (int s = 64; s > 0; s >>= 1) {
        if (t < s) red[t] += red[t + s];
        __syncthreads();
    }
    const float mu = red[0] * (1.0f / DD);
    __syncthreads();
    const float dy = y - mu;
    red[t] = dy * dy;
    __syncthreads();
    for (int s = 64; s > 0; s >>= 1) {
        if (t < s) red[t] += red[t + s];
        __syncthreads();
    }
    const float var = red[0] * (1.0f / DD);
    out[(size_t)node * DD + t] = dy * rsqrtf(var + LN_EPS) * gamma[t] + beta[t];
}

// ---------------------------------------------------------------------------
extern "C" void kernel_launch(void* const* d_in, const int* in_sizes, int n_in,
                              void* d_out, int out_size, void* d_ws, size_t ws_size,
                              hipStream_t stream)
{
    const float* x        = (const float*)d_in[0];
    const float* edge0    = (const float*)d_in[1];
    const int*   edge_mask= (const int*)  d_in[2];
    const float* W1m      = (const float*)d_in[3];
    const float* b1m      = (const float*)d_in[4];
    const float* W2m      = (const float*)d_in[5];
    const float* b2m      = (const float*)d_in[6];
    const float* W1u      = (const float*)d_in[7];
    const float* b1u      = (const float*)d_in[8];
    const float* W2u      = (const float*)d_in[9];
    const float* b2u      = (const float*)d_in[10];
    const float* gamma    = (const float*)d_in[11];
    const float* beta     = (const float*)d_in[12];
    float*       out      = (float*)d_out;

    const int nodes = BB * NN;                 // 2048
    float* Apre = (float*)d_ws;                // nodes*DD floats (1 MB)
    float* Bp   = Apre + (size_t)nodes * DD;   // 1 MB
    float* hs   = Bp   + (size_t)nodes * DD;   // 1 MB
    (void)in_sizes; (void)n_in; (void)out_size; (void)ws_size;

    ean_phase1<<<nodes, DD,  0, stream>>>(x, W1m, b1m, Apre, Bp);
    ean_phase2<<<nodes, 256, 0, stream>>>(edge0, edge_mask, W1m, Apre, Bp, hs);
    ean_phase3<<<nodes, DD,  0, stream>>>(x, edge_mask, hs, W2m, b2m,
                                          W1u, b1u, W2u, b2u, gamma, beta, out);
}
